// GQA_91087666414115
// MI455X (gfx1250) — matrix operand order repaired and checked by reading, not compiled, not run on hardware
//
#include <hip/hip_runtime.h>
#include <hip/hip_bf16.h>

// ---------------------------------------------------------------------------
// GQA attention (RoPE + causal softmax + GQA) for MI455X / gfx1250, wave32.
// - All contractions via v_wmma_f32_16x16x32_f16 (f32 accumulate).
// - Weights pre-transposed so GEMM B fragments are contiguous global b128s
//   (no LDS in GEMM; B stays resident in the 192MB L2).
// - Attention K/V tiles staged into LDS with GLOBAL_LOAD_ASYNC_TO_LDS_B128
//   (ASYNCcnt path), V pre-transposed to [B,NKV,DK,S] by the pack kernel.
// ---------------------------------------------------------------------------

typedef __attribute__((ext_vector_type(16))) _Float16 v16h;
typedef __attribute__((ext_vector_type(8)))  float    v8f;

#define B_   4
#define S_   2048
#define D_   1024
#define NH_  16
#define NKV_ 4
#define DK_  64
#define NG_  4   // NH / NKV

static __device__ __forceinline__ v8f wmma_f16(v16h a, v16h b, v8f c) {
  return __builtin_amdgcn_wmma_f32_16x16x32_f16(
      /*neg_a=*/false, a, /*neg_b=*/false, b,
      /*c_mod=*/(short)0, c, /*reuse_a=*/false, /*reuse_b=*/false);
}

// A fragment (16x32 f16, M x K). base points at (row0, k0). Contiguous order:
// lanes 0-15 hold row M=l with K 0..15; lanes 16-31 K 16..31.
static __device__ __forceinline__ v16h load_a_frag(const _Float16* base, int lda, int lane) {
  int l  = lane & 15;
  int hi = lane >> 4;
  const _Float16* p = base + (long)l * lda + hi * 16;
  v16h a;
#pragma unroll
  for (int i = 0; i < 8; ++i) a[i]     = p[i];
#pragma unroll
  for (int i = 0; i < 8; ++i) a[i + 8] = p[8 + i];
  return a;
}

// B fragment (32x16 f16, K x N) from N-major storage Bt[n][k] (ld = ldk).
// Lane l<16: column n0+l, K 0..15; lanes 16-31: column n0+l, K 16..31.
static __device__ __forceinline__ v16h load_b_frag(const _Float16* base, int ldk, int lane) {
  int l  = lane & 15;
  int hi = lane >> 4;
  const _Float16* p = base + (long)l * ldk + hi * 16;
  v16h b;
#pragma unroll
  for (int i = 0; i < 16; ++i) b[i] = p[i];
  return b;
}

// gfx1250 async copy: 16 bytes global -> LDS, tracked by ASYNCcnt.
// VDST = VGPR with LDS byte address, VADDR = 64-bit global address, SADDR=off.
static __device__ __forceinline__ void async_copy_b128(unsigned lds_addr,
                                                       const _Float16* gsrc) {
  unsigned long long ga = (unsigned long long)(const void*)gsrc;
  asm volatile("global_load_async_to_lds_b128 %0, %1, off"
               :: "v"(lds_addr), "v"(ga) : "memory");
}
static __device__ __forceinline__ void async_wait0() {
  asm volatile("s_wait_asynccnt 0x0" ::: "memory");
}

// ---------------------------------------------------------------------------
__global__ void f32_to_f16_kernel(const float* __restrict__ in,
                                  _Float16* __restrict__ out, long n) {
  long i      = (long)blockIdx.x * blockDim.x + threadIdx.x;
  long stride = (long)gridDim.x * blockDim.x;
  for (; i < n; i += stride) out[i] = (_Float16)in[i];
}

// W [K,N] f32 -> Wt [N,K] f16 (one-time; lets GEMM read B fragments directly).
__global__ void f32_to_f16_transpose_kernel(const float* __restrict__ in,
                                            _Float16* __restrict__ out,
                                            int K, int N) {
  long i      = (long)blockIdx.x * blockDim.x + threadIdx.x;
  long total  = (long)K * N;
  long stride = (long)gridDim.x * blockDim.x;
  for (; i < total; i += stride) {
    int n = (int)(i % N);
    int k = (int)(i / N);
    out[(long)n * K + k] = (_Float16)in[i];
  }
}

// ---------------------------------------------------------------------------
// C[M,N] = A[M,K] (f16 row-major) x Bt^T, Bt = [N,K] f16 row-major -> f32.
// Block tile 64x64, 128 threads = 4 waves; wave w owns rows [w*16, +16).
// No LDS: A and B fragments load straight from global (B is L2-resident).
__global__ __launch_bounds__(128)
void gemm_f16_wmma_kernel(const _Float16* __restrict__ A,
                          const _Float16* __restrict__ Bt,
                          float* __restrict__ C,
                          int M, int N, int K) {
  int n0   = blockIdx.x * 64;
  int m0   = blockIdx.y * 64;
  int wave = threadIdx.x >> 5, lane = threadIdx.x & 31;
  int mw   = m0 + wave * 16;
  int l    = lane & 15, hi = lane >> 4;

  const _Float16* arow = A + (long)mw * K;
  v8f acc[4] = {};

  for (int kb = 0; kb < K; kb += 32) {
    v16h a = load_a_frag(arow + kb, K, lane);
    if (kb + 32 < K) {   // global_prefetch_b8 for the next k-slab
      __builtin_prefetch(arow + (long)l * K + kb + 32, 0, 3);
      __builtin_prefetch(Bt + (long)(n0 + l * 4) * K + kb + 32, 0, 3);
    }
#pragma unroll
    for (int j = 0; j < 4; ++j) {
      v16h b = load_b_frag(Bt + (long)(n0 + j * 16) * K + kb, K, lane);
      acc[j] = wmma_f16(a, b, acc[j]);
    }
  }

#pragma unroll
  for (int j = 0; j < 4; ++j)
#pragma unroll
    for (int r = 0; r < 8; ++r)
      C[(long)(mw + r + hi * 8) * N + n0 + j * 16 + l] = acc[j][r];
}

// ---------------------------------------------------------------------------
// RoPE + layout pack: raw [B,S,H*DK] f32 -> f16, either
//   transpose_out=0: out [B,H,S,DK]   (Q, K)
//   transpose_out=1: out [B,H,DK,S]   (V; d-major so attention V tiles copy flat)
__global__ void rope_pack_kernel(const float* __restrict__ raw,
                                 _Float16* __restrict__ out,
                                 int H, int do_rope, int transpose_out) {
  long idx   = (long)blockIdx.x * blockDim.x + threadIdx.x;
  long total = (long)B_ * S_ * H * (DK_ / 2);
  if (idx >= total) return;
  int  d = (int)(idx % (DK_ / 2));
  long t = idx / (DK_ / 2);
  int  h = (int)(t % H); t /= H;
  int  s = (int)(t % S_); t /= S_;
  int  b = (int)t;

  const float* rp = raw + (((long)b * S_ + s) * H + h) * DK_;
  float x1 = rp[d], x2 = rp[d + 32];
  float o1 = x1, o2 = x2;
  if (do_rope) {
    float theta = __expf(-0.28782313f * (float)d);  // 10000^(-d/32)
    float ang   = (float)s * theta;
    float sn, cs;
    __sincosf(ang, &sn, &cs);
    o1 = x1 * cs - x2 * sn;
    o2 = x2 * cs + x1 * sn;
  }
  if (transpose_out) {
    _Float16* op = out + (((long)b * H + h) * DK_) * S_;
    op[(long)d * S_ + s]        = (_Float16)o1;
    op[(long)(d + 32) * S_ + s] = (_Float16)o2;
  } else {
    _Float16* op = out + (((long)b * H + h) * S_ + s) * DK_;
    op[d]      = (_Float16)o1;
    op[d + 32] = (_Float16)o2;
  }
}

// ---------------------------------------------------------------------------
// Flash-style causal GQA attention. Block = (b,h, 64-row q tile), 128 threads.
// Wave w owns q rows [q0+w*16, +16). K tile [kk][d] and V tile [d][kk] staged
// with async b128 copies; online softmax via __shfl_xor across the 16 lanes
// that share an accumulator row. Output written [B,S,NH*DK] f16.
__global__ __launch_bounds__(128)
void attn_kernel(const _Float16* __restrict__ qh,
                 const _Float16* __restrict__ kh,
                 const _Float16* __restrict__ vt,
                 _Float16* __restrict__ oh) {
  __shared__ _Float16 Kt[64 * 64];        // [kk][d]
  __shared__ _Float16 Vt[64 * 64];        // [d][kk]
  __shared__ _Float16 Pw[4][16 * 64];     // per-wave P scratch [m][kk]

  int bh  = blockIdx.y;
  int b   = bh / NH_, h = bh % NH_;
  int kvh = h / NG_;
  int q0  = blockIdx.x * 64;
  int tid = threadIdx.x, wave = tid >> 5, lane = tid & 31;
  int l   = lane & 15, hi = lane >> 4;
  int mq  = q0 + wave * 16;

  const _Float16* qbase = qh + (((long)b * NH_ + h) * S_ + mq) * DK_;
  v16h qf0 = load_a_frag(qbase,      DK_, lane);
  v16h qf1 = load_a_frag(qbase + 32, DK_, lane);

  const _Float16* kbase = kh + ((long)b * NKV_ + kvh) * S_ * DK_;   // [s][d]
  const _Float16* vbase = vt + ((long)b * NKV_ + kvh) * DK_ * S_;   // [d][s]

  unsigned ktb = (unsigned)(size_t)(void*)Kt;   // LDS byte addresses
  unsigned vtb = (unsigned)(size_t)(void*)Vt;
  int dV  = tid >> 1;             // V tile row (d) handled by this thread
  int cbV = (tid & 1) * 32;       // V tile col base

  float mrow[8], lrow[8];
  v8f   oacc[4] = {};
#pragma unroll
  for (int r = 0; r < 8; ++r) { mrow[r] = -3.0e38f; lrow[r] = 0.f; }
  const float scale = 0.125f;  // 1/sqrt(64)

  for (int kb = 0; kb <= q0; kb += 64) {
    // --- async-stage K (flat copy) and V (row-strided copy) tiles ---
    const _Float16* ksrc = kbase + (long)kb * DK_;
    const _Float16* vsrc = vbase + kb;
#pragma unroll
    for (int i = 0; i < 4; ++i) {
      async_copy_b128(ktb + tid * 64 + i * 16, ksrc + tid * 32 + i * 8);
      async_copy_b128(vtb + tid * 64 + i * 16,
                      vsrc + (long)dV * S_ + cbV + i * 8);
    }
    async_wait0();
    __syncthreads();

    // --- scores S = Q K^T (16 x 64 per wave) ---
    float p[4][8];
    float rmax[8];
#pragma unroll
    for (int r = 0; r < 8; ++r) rmax[r] = -3.0e38f;
#pragma unroll
    for (int j = 0; j < 4; ++j) {
      v8f  s   = {};
      v16h bk0 = load_b_frag(&Kt[(j * 16) * 64],      64, lane);
      v16h bk1 = load_b_frag(&Kt[(j * 16) * 64 + 32], 64, lane);
      s = wmma_f16(qf0, bk0, s);
      s = wmma_f16(qf1, bk1, s);
      int n = kb + j * 16 + l;  // key position of this lane's column
#pragma unroll
      for (int r = 0; r < 8; ++r) {
        int   m  = mq + r + hi * 8;                     // query position
        float sv = (n <= m) ? s[r] * scale : -3.0e38f;  // causal mask
        p[j][r]  = sv;
        rmax[r]  = fmaxf(rmax[r], sv);
      }
    }

    // --- online softmax (row = 16 lanes of a half-wave) ---
#pragma unroll
    for (int r = 0; r < 8; ++r) {
#pragma unroll
      for (int off = 1; off < 16; off <<= 1)
        rmax[r] = fmaxf(rmax[r], __shfl_xor(rmax[r], off, 32));
      float mnew = fmaxf(mrow[r], rmax[r]);
      float corr = __expf(mrow[r] - mnew);
      mrow[r] = mnew;
      lrow[r] *= corr;
#pragma unroll
      for (int j = 0; j < 4; ++j) oacc[j][r] *= corr;
      float rsum = 0.f;
#pragma unroll
      for (int j = 0; j < 4; ++j) {
        float e = __expf(p[j][r] - mnew);
        p[j][r] = e;
        rsum += e;
      }
#pragma unroll
      for (int off = 1; off < 16; off <<= 1)
        rsum += __shfl_xor(rsum, off, 32);
      lrow[r] += rsum;
    }

    // --- reshape P (C layout) -> A layout via per-wave LDS patch ---
#pragma unroll
    for (int j = 0; j < 4; ++j)
#pragma unroll
      for (int r = 0; r < 8; ++r)
        Pw[wave][(r + hi * 8) * 64 + j * 16 + l] = (_Float16)p[j][r];
    __syncthreads();

    v16h pa0 = load_a_frag(&Pw[wave][0],  64, lane);
    v16h pa1 = load_a_frag(&Pw[wave][32], 64, lane);
#pragma unroll
    for (int j = 0; j < 4; ++j) {
      v16h bv0 = load_b_frag(&Vt[(j * 16) * 64],      64, lane);
      v16h bv1 = load_b_frag(&Vt[(j * 16) * 64 + 32], 64, lane);
      oacc[j] = wmma_f16(pa0, bv0, oacc[j]);
      oacc[j] = wmma_f16(pa1, bv1, oacc[j]);
    }
    __syncthreads();
  }

  // --- epilogue: normalize, write [B,S,NH*DK] f16 ---
#pragma unroll
  for (int r = 0; r < 8; ++r) {
    float inv = 1.0f / lrow[r];
    int   m   = mq + r + hi * 8;
    _Float16* op = oh + ((long)b * S_ + m) * D_ + h * DK_;
#pragma unroll
    for (int j = 0; j < 4; ++j)
      op[j * 16 + l] = (_Float16)(oacc[j][r] * inv);
  }
}

// ---------------------------------------------------------------------------
extern "C" void kernel_launch(void* const* d_in, const int* in_sizes, int n_in,
                              void* d_out, int out_size, void* d_ws, size_t ws_size,
                              hipStream_t stream) {
  (void)in_sizes; (void)n_in; (void)out_size; (void)ws_size;
  const float* x  = (const float*)d_in[0];
  const float* Wq = (const float*)d_in[1];
  const float* Wk = (const float*)d_in[2];
  const float* Wv = (const float*)d_in[3];
  const float* Wo = (const float*)d_in[4];
  // d_in[5] = mask (unused; causal mask computed analytically)

  char* ws = (char*)d_ws;
  // workspace layout (bytes); weights stored TRANSPOSED [N,K] f16
  _Float16* xh   = (_Float16*)(ws + 0);          // 16 MB  [B,S,D] f16
  _Float16* wqt  = (_Float16*)(ws + 16777216);   //  2 MB  [D,D]^T
  _Float16* wkt  = (_Float16*)(ws + 18874368);   // .5 MB  [256,D]
  _Float16* wvt  = (_Float16*)(ws + 19398656);   // .5 MB  [256,D]
  _Float16* wot  = (_Float16*)(ws + 19922944);   //  2 MB  [D,D]^T
  float*    qraw = (float*)   (ws + 22020096);   // 32 MB  [B,S,NH*DK] f32
  float*    kraw = (float*)   (ws + 55574528);   //  8 MB
  float*    vraw = (float*)   (ws + 63963136);   //  8 MB
  _Float16* qh16 = (_Float16*)(ws + 72351744);   // 16 MB  [B,NH,S,DK]
  _Float16* kh16 = (_Float16*)(ws + 89128960);   //  4 MB  [B,NKV,S,DK]
  _Float16* vt16 = (_Float16*)(ws + 93323264);   //  4 MB  [B,NKV,DK,S]
  _Float16* oh   = (_Float16*)(ws + 22020096);   // reuse qraw: 16 MB [B,S,D] f16

  const int BS = B_ * S_;  // 8192 rows

  // 1) f32 -> f16 conversions (weights transposed)
  f32_to_f16_kernel<<<4096, 256, 0, stream>>>(x, xh, (long)BS * D_);
  f32_to_f16_transpose_kernel<<<2048, 256, 0, stream>>>(Wq, wqt, D_, D_);
  f32_to_f16_transpose_kernel<<<512,  256, 0, stream>>>(Wk, wkt, D_, NKV_ * DK_);
  f32_to_f16_transpose_kernel<<<512,  256, 0, stream>>>(Wv, wvt, D_, NKV_ * DK_);
  f32_to_f16_transpose_kernel<<<2048, 256, 0, stream>>>(Wo, wot, D_, D_);

  // 2) QKV projections (WMMA GEMMs, B from transposed weights)
  gemm_f16_wmma_kernel<<<dim3(D_ / 64, BS / 64), 128, 0, stream>>>(
      xh, wqt, qraw, BS, D_, D_);
  gemm_f16_wmma_kernel<<<dim3((NKV_ * DK_) / 64, BS / 64), 128, 0, stream>>>(
      xh, wkt, kraw, BS, NKV_ * DK_, D_);
  gemm_f16_wmma_kernel<<<dim3((NKV_ * DK_) / 64, BS / 64), 128, 0, stream>>>(
      xh, wvt, vraw, BS, NKV_ * DK_, D_);

  // 3) RoPE + repack (V transposed to d-major for flat async staging)
  rope_pack_kernel<<<(B_ * S_ * NH_  * (DK_ / 2)) / 256, 256, 0, stream>>>(qraw, qh16, NH_,  1, 0);
  rope_pack_kernel<<<(B_ * S_ * NKV_ * (DK_ / 2)) / 256, 256, 0, stream>>>(kraw, kh16, NKV_, 1, 0);
  rope_pack_kernel<<<(B_ * S_ * NKV_ * (DK_ / 2)) / 256, 256, 0, stream>>>(vraw, vt16, NKV_, 0, 1);

  // 4) causal GQA attention -> oh [B,S,NH*DK] f16
  attn_kernel<<<dim3(S_ / 64, B_ * NH_), 128, 0, stream>>>(qh16, kh16, vt16, oh);

  // 5) output projection -> d_out f32
  gemm_f16_wmma_kernel<<<dim3(D_ / 64, BS / 64), 128, 0, stream>>>(
      oh, wot, (float*)d_out, BS, D_, D_);
}
